// Qwen2_89988154785929
// MI455X (gfx1250) — matrix operand order repaired and checked by reading, not compiled, not run on hardware
//
#include <hip/hip_runtime.h>

// ---------------- model dimensions (match reference) ----------------
constexpr int NL   = 4;
constexpr int SEQ  = 2048;
constexpr int DM   = 1024;
constexpr int NHQ  = 16;
constexpr int NKV  = 4;
constexpr int HDIM = 64;
constexpr int GRP  = NHQ / NKV;      // 4
constexpr int FF   = 4096;
constexpr int VOC  = 32000;
constexpr int QSTR = NHQ * HDIM;     // 1024
constexpr int KSTR = NKV * HDIM;     // 256

typedef __attribute__((ext_vector_type(16))) __bf16 v16bf;
typedef __attribute__((ext_vector_type(8)))  float  v8f;
typedef unsigned int u32x4 __attribute__((ext_vector_type(4)));
typedef int          i32x4 __attribute__((ext_vector_type(4)));
typedef int          i32x8 __attribute__((ext_vector_type(8)));

#if defined(__has_builtin)
#  if __has_builtin(__builtin_amdgcn_tensor_load_to_lds) && __has_builtin(__builtin_amdgcn_s_wait_tensorcnt)
#    define USE_TDM 1
#  endif
#endif
#ifndef USE_TDM
#  define USE_TDM 0
#endif

// ---------------- helpers ----------------
__device__ __forceinline__ __bf16 f2bf(float f) {
  unsigned u = __builtin_bit_cast(unsigned, f);
  unsigned r = (u + 0x7fffu + ((u >> 16) & 1u)) >> 16;   // round-to-nearest-even
  return __builtin_bit_cast(__bf16, (unsigned short)r);
}
__device__ __forceinline__ v8f zero8() {
  v8f z;
#pragma unroll
  for (int i = 0; i < 8; ++i) z[i] = 0.0f;
  return z;
}
union FragU { uint4 q[2]; v16bf v; };
__device__ __forceinline__ v16bf ld_frag(const __bf16* p0, const __bf16* p1) {
  FragU f;
  f.q[0] = *(const uint4*)p0;   // ds_load_b128
  f.q[1] = *(const uint4*)p1;   // ds_load_b128
  return f.v;
}

// ---------------- weight convert + transpose: src[K][N] f32 -> dst[N][K] bf16 ----------------
__global__ __launch_bounds__(256) void convT_kernel(const float* __restrict__ src,
                                                    __bf16* __restrict__ dst,
                                                    int K, int N) {
  size_t idx = (size_t)blockIdx.x * 256 + threadIdx.x;
  if (idx >= (size_t)K * N) return;
  int n = (int)(idx / (size_t)K);
  int k = (int)(idx % (size_t)K);
  dst[idx] = f2bf(src[(size_t)k * N + n]);
}

// ---------------- plain f32 -> bf16 convert ----------------
__global__ __launch_bounds__(256) void conv_kernel(const float* __restrict__ src,
                                                   __bf16* __restrict__ dst, int n) {
  int i = blockIdx.x * 256 + threadIdx.x;
  if (i < n) dst[i] = f2bf(src[i]);
}

// ---------------- embedding gather ----------------
__global__ __launch_bounds__(256) void gather_kernel(const float* __restrict__ embed,
                                                     const int* __restrict__ ids,
                                                     float* __restrict__ h) {
  int idx = blockIdx.x * 256 + threadIdx.x;        // SEQ*DM threads
  int s = idx >> 10, d = idx & (DM - 1);
  h[idx] = embed[(size_t)ids[s] * DM + d];
}

// ---------------- RMSNorm (f32 in, bf16 out), one row per block ----------------
__global__ __launch_bounds__(256) void rmsnorm_kernel(const float* __restrict__ x,
                                                      const float* __restrict__ w,
                                                      __bf16* __restrict__ out) {
  __shared__ float red[8];
  int row = blockIdx.x, t = threadIdx.x;
  const float* xr = x + (size_t)row * DM;
  float s = 0.0f;
  for (int i = t; i < DM; i += 256) { float v = xr[i]; s += v * v; }
#pragma unroll
  for (int off = 1; off < 32; off <<= 1) s += __shfl_xor(s, off, 32);
  if ((t & 31) == 0) red[t >> 5] = s;
  __syncthreads();
  float tot = 0.0f;
#pragma unroll
  for (int i = 0; i < 8; ++i) tot += red[i];
  float inv = rsqrtf(tot * (1.0f / DM) + 1e-6f);
  for (int i = t; i < DM; i += 256)
    out[(size_t)row * DM + i] = f2bf(w[i] * xr[i] * inv);
}

// ---------------- RoPE: f32 in -> roped bf16 out (pair (i, i+32) per thread) ----------------
__global__ __launch_bounds__(256) void rope_bf16_kernel(const float* __restrict__ src,
                                                        __bf16* __restrict__ dst, int nheads) {
  int idx = blockIdx.x * 256 + threadIdx.x;   // SEQ*nheads*32 threads
  int per = nheads * 32;
  int s = idx / per, rem = idx % per;
  int hh = rem >> 5, i = rem & 31;
  float inv = __expf(-((float)(2 * i) / 64.0f) * 9.210340372f);  // 10000^(-2i/64)
  float ang = (float)s * inv;
  float c, sn;
  __sincosf(ang, &sn, &c);
  size_t base = (size_t)s * nheads * HDIM + hh * HDIM;
  float x1 = src[base + i], x2 = src[base + i + 32];
  dst[base + i]      = f2bf(x1 * c - x2 * sn);
  dst[base + i + 32] = f2bf(x2 * c + x1 * sn);
}

// ---------------- SiLU(g) * u -> bf16 ----------------
__global__ __launch_bounds__(256) void silu_mul_kernel(const float* __restrict__ g,
                                                       const float* __restrict__ u,
                                                       __bf16* __restrict__ o, int n) {
  int i = blockIdx.x * 256 + threadIdx.x;
  if (i < n) {
    float gv = g[i];
    o[i] = f2bf(gv / (1.0f + __expf(-gv)) * u[i]);
  }
}

// ---------------- WMMA bf16 GEMM: C[M,N](f32) = A[M,K](bf16) @ Bt[N,K](bf16)^T ----------------
// Block tile 128x64, BK=64. 8 waves in 4(M)x2(N); each wave 32x32 via 2x2 register block.
// All tile movement by the Tensor Data Mover, double-buffered in LDS:
// wave 0 issues the next (A,B) tile pair, s_wait_tensorcnt 2 retires the current pair,
// so DMA of tile t+1 overlaps the 8 WMMAs of tile t.
__global__ __launch_bounds__(256) void gemm_bf16_wmma(const __bf16* __restrict__ A,
                                                      const __bf16* __restrict__ Bt,
                                                      const float* __restrict__ bias,
                                                      const float* __restrict__ R,
                                                      float* __restrict__ C,
                                                      int M, int N, int K) {
  __shared__ alignas(16) __bf16 As[2][128][72];  // [buf][m][k], +8 pad (TDM pad: 4dw per 32dw)
  __shared__ alignas(16) __bf16 Bs[2][64][64];   // [buf][n][k], contiguous

  const int tid  = threadIdx.x;
  const int lane = tid & 31;
  const int w    = tid >> 5;
  const int wm   = w >> 1;            // 0..3
  const int wn   = w & 1;             // 0..1
  const int rlo  = lane & 15;
  const int rhi  = lane >> 4;
  const int m0   = blockIdx.y * 128;
  const int n0   = blockIdx.x * 64;
  const int nt   = K >> 6;

  v8f acc[2][2];
#pragma unroll
  for (int i = 0; i < 2; ++i)
#pragma unroll
    for (int j = 0; j < 2; ++j) acc[i][j] = zero8();

  auto compute = [&](int buf) {
#pragma unroll
    for (int ks = 0; ks < 2; ++ks) {
      v16bf a0 = ld_frag(&As[buf][wm * 32 + rlo][ks * 32 + rhi * 8],
                         &As[buf][wm * 32 + rlo][ks * 32 + 16 + rhi * 8]);
      v16bf a1 = ld_frag(&As[buf][wm * 32 + 16 + rlo][ks * 32 + rhi * 8],
                         &As[buf][wm * 32 + 16 + rlo][ks * 32 + 16 + rhi * 8]);
      v16bf b0 = ld_frag(&Bs[buf][wn * 32 + rlo][ks * 32 + rhi * 8],
                         &Bs[buf][wn * 32 + rlo][ks * 32 + 16 + rhi * 8]);
      v16bf b1 = ld_frag(&Bs[buf][wn * 32 + 16 + rlo][ks * 32 + rhi * 8],
                         &Bs[buf][wn * 32 + 16 + rlo][ks * 32 + 16 + rhi * 8]);
      acc[0][0] = __builtin_amdgcn_wmma_f32_16x16x32_bf16(false, a0, false, b0, (short)0, acc[0][0], false, false);
      acc[0][1] = __builtin_amdgcn_wmma_f32_16x16x32_bf16(false, a0, false, b1, (short)0, acc[0][1], false, false);
      acc[1][0] = __builtin_amdgcn_wmma_f32_16x16x32_bf16(false, a1, false, b0, (short)0, acc[1][0], false, false);
      acc[1][1] = __builtin_amdgcn_wmma_f32_16x16x32_bf16(false, a1, false, b1, (short)0, acc[1][1], false, false);
    }
  };

#if USE_TDM
  typedef __attribute__((address_space(3))) __bf16 lds_bf16_t;
  auto issue_pair = [&](int k0, int buf) {
    if (tid < 32) {   // wave 0 drives the Tensor Data Mover
      i32x4 z4 = {0, 0, 0, 0};
#if __clang_major__ >= 23
      i32x8 z8 = {0, 0, 0, 0, 0, 0, 0, 0};
#endif
      {   // A tile: 128 rows x 64 k, LDS-padded to [128][72]
        unsigned long long ga = (unsigned long long)(uintptr_t)(A + (size_t)m0 * K + k0);
        unsigned ldsoff = (unsigned)(size_t)(lds_bf16_t*)&As[buf][0][0];
        u32x4 g0;
        g0.x = 1u;
        g0.y = ldsoff;
        g0.z = (unsigned)ga;
        g0.w = (unsigned)((ga >> 32) & 0x1FFFFFFu) | (2u << 30);
        i32x8 g1;
        g1[0] = (int)((1u << 16) | (1u << 20) | (4u << 22) | (3u << 25)); // 2B; pad 4dw per 32dw
        g1[1] = (int)(((unsigned)K & 0xFFFFu) << 16);                      // tensor_dim0 lo16
        g1[2] = (int)((((unsigned)K >> 16) & 0xFFFFu) | (((unsigned)M & 0xFFFFu) << 16));
        g1[3] = (int)((((unsigned)M >> 16) & 0xFFFFu) | (64u << 16));      // tile_dim0=64
        g1[4] = (int)128u;                                                 // tile_dim1=128
        g1[5] = (int)(unsigned)K;                                          // dim0 stride
        g1[6] = 0; g1[7] = 0;
#if __clang_major__ >= 23
        __builtin_amdgcn_tensor_load_to_lds(g0, g1, z4, z4, z8, 0);
#else
        __builtin_amdgcn_tensor_load_to_lds(g0, g1, z4, z4, 0);
#endif
      }
      {   // B tile: 64 rows x 64 k, contiguous LDS
        unsigned long long ga = (unsigned long long)(uintptr_t)(Bt + (size_t)n0 * K + k0);
        unsigned ldsoff = (unsigned)(size_t)(lds_bf16_t*)&Bs[buf][0][0];
        u32x4 g0;
        g0.x = 1u;
        g0.y = ldsoff;
        g0.z = (unsigned)ga;
        g0.w = (unsigned)((ga >> 32) & 0x1FFFFFFu) | (2u << 30);
        i32x8 g1;
        g1[0] = (int)(1u << 16);                                           // 2B elements
        g1[1] = (int)(((unsigned)K & 0xFFFFu) << 16);
        g1[2] = (int)((((unsigned)K >> 16) & 0xFFFFu) | (((unsigned)N & 0xFFFFu) << 16));
        g1[3] = (int)((((unsigned)N >> 16) & 0xFFFFu) | (64u << 16));      // tile_dim0=64
        g1[4] = (int)64u;                                                  // tile_dim1=64
        g1[5] = (int)(unsigned)K;
        g1[6] = 0; g1[7] = 0;
#if __clang_major__ >= 23
        __builtin_amdgcn_tensor_load_to_lds(g0, g1, z4, z4, z8, 0);
#else
        __builtin_amdgcn_tensor_load_to_lds(g0, g1, z4, z4, 0);
#endif
      }
    }
  };

  issue_pair(0, 0);
  for (int t = 0; t < nt; ++t) {
    const int buf = t & 1;
    if (t + 1 < nt) {
      issue_pair((t + 1) * 64, buf ^ 1);       // overlap DMA of next tile with compute
      if (tid < 32) __builtin_amdgcn_s_wait_tensorcnt(2);  // current pair retired
    } else {
      if (tid < 32) __builtin_amdgcn_s_wait_tensorcnt(0);
    }
    __syncthreads();
    compute(buf);
    __syncthreads();
  }
#else
  const int arow = tid >> 1;          // 0..127
  const int aseg = (tid & 1) * 32;    // 0 or 32 (bf16 elements)
  const int brow = tid >> 2;          // 0..63
  const int bseg = (tid & 3) * 16;    // 0,16,32,48
  for (int t = 0; t < nt; ++t) {
    const int buf = t & 1;
    const int k0 = t * 64;
    const uint4* ap = (const uint4*)(A + (size_t)(m0 + arow) * K + k0 + aseg);
    __builtin_prefetch((const void*)(A + (size_t)(m0 + arow) * K + k0 + 64 + aseg), 0, 1);
    uint4 a0 = ap[0], a1 = ap[1], a2 = ap[2], a3 = ap[3];
    *(uint4*)&As[buf][arow][aseg + 0]  = a0;
    *(uint4*)&As[buf][arow][aseg + 8]  = a1;
    *(uint4*)&As[buf][arow][aseg + 16] = a2;
    *(uint4*)&As[buf][arow][aseg + 24] = a3;
    const uint4* bp = (const uint4*)(Bt + (size_t)(n0 + brow) * K + k0 + bseg);
    uint4 b0 = bp[0], b1 = bp[1];
    *(uint4*)&Bs[buf][brow][bseg + 0] = b0;
    *(uint4*)&Bs[buf][brow][bseg + 8] = b1;
    __syncthreads();
    compute(buf);
    __syncthreads();
  }
#endif

#pragma unroll
  for (int i = 0; i < 2; ++i)
#pragma unroll
    for (int j = 0; j < 2; ++j)
#pragma unroll
      for (int r = 0; r < 8; ++r) {
        int row = m0 + wm * 32 + i * 16 + rhi * 8 + r;
        int col = n0 + wn * 32 + j * 16 + rlo;
        float vv = acc[i][j][r];
        if (bias) vv += bias[col];
        if (R)    vv += R[(size_t)row * N + col];
        C[(size_t)row * N + col] = vv;
      }
}

// ---------------- flash attention: 4 independent waves per block, 16-query tile each ----------
// bf16 inputs (roped Q/K, converted V) -> b128 vector tile loads; scale applied on f32 scores.
__global__ __launch_bounds__(128) void attn_kernel(const __bf16* __restrict__ Q,
                                                   const __bf16* __restrict__ Kc,
                                                   const __bf16* __restrict__ Vc,
                                                   __bf16* __restrict__ O) {
  __shared__ alignas(16) __bf16 Qs[4][16][72];   // [wave][q][hd]
  __shared__ alignas(16) __bf16 Ks[4][32][72];   // [wave][key][hd]
  __shared__ alignas(16) __bf16 Vt[4][64][40];   // [wave][hd][key] (transposed)
  __shared__ alignas(16) __bf16 Ps[4][16][40];   // [wave][q][key]

  const int tid  = threadIdx.x;
  const int lane = tid & 31;
  const int w    = tid >> 5;                     // 0..3
  const int qt   = blockIdx.x * 4 + w;
  const int h    = blockIdx.y;
  const int kvh  = h / GRP;
  const int q0   = qt * 16;
  const int rlo  = lane & 15, rhi = lane >> 4;
  const float scale = 0.125f;                    // 1/sqrt(64)

  for (int i = lane; i < 16 * 8; i += 32) {      // 16 rows x 8 b128 chunks
    int r = i >> 3, seg = (i & 7) * 8;
    *(uint4*)&Qs[w][r][seg] = *(const uint4*)(Q + (size_t)(q0 + r) * QSTR + h * HDIM + seg);
  }
  v16bf qa0 = ld_frag(&Qs[w][rlo][rhi * 16],      &Qs[w][rlo][rhi * 16 + 8]);      // hd 0..31
  v16bf qa1 = ld_frag(&Qs[w][rlo][32 + rhi * 16], &Qs[w][rlo][32 + rhi * 16 + 8]); // hd 32..63

  v8f acc[4];
#pragma unroll
  for (int c = 0; c < 4; ++c) acc[c] = zero8();
  float m[8], l[8];
#pragma unroll
  for (int r = 0; r < 8; ++r) { m[r] = -1e30f; l[r] = 0.0f; }

  const int nkt = ((q0 + 15) >> 5) + 1;          // causal tiles of 32 keys
  for (int t = 0; t < nkt; ++t) {
    const int kb = t * 32;
    for (int i = lane; i < 32 * 8; i += 32) {    // K tile: vector copy
      int r = i >> 3, seg = (i & 7) * 8;
      *(uint4*)&Ks[w][r][seg] = *(const uint4*)(Kc + (size_t)(kb + r) * KSTR + kvh * HDIM + seg);
    }
    for (int i = lane; i < 32 * 8; i += 32) {    // V tile: vector load, transposed scatter
      int r = i >> 3, seg = (i & 7) * 8;
      uint4 vv = *(const uint4*)(Vc + (size_t)(kb + r) * KSTR + kvh * HDIM + seg);
      const __bf16* e = (const __bf16*)&vv;
#pragma unroll
      for (int j = 0; j < 8; ++j) Vt[w][seg + j][r] = e[j];
    }

    // scores: two 16x16 key tiles, hd-K of 64 in 2 wmma steps each
    v8f s0 = zero8(), s1 = zero8();
    {
      v16bf b00 = ld_frag(&Ks[w][rlo][rhi * 16],           &Ks[w][rlo][rhi * 16 + 8]);
      v16bf b01 = ld_frag(&Ks[w][rlo][32 + rhi * 16],      &Ks[w][rlo][32 + rhi * 16 + 8]);
      v16bf b10 = ld_frag(&Ks[w][16 + rlo][rhi * 16],      &Ks[w][16 + rlo][rhi * 16 + 8]);
      v16bf b11 = ld_frag(&Ks[w][16 + rlo][32 + rhi * 16], &Ks[w][16 + rlo][32 + rhi * 16 + 8]);
      s0 = __builtin_amdgcn_wmma_f32_16x16x32_bf16(false, qa0, false, b00, (short)0, s0, false, false);
      s0 = __builtin_amdgcn_wmma_f32_16x16x32_bf16(false, qa1, false, b01, (short)0, s0, false, false);
      s1 = __builtin_amdgcn_wmma_f32_16x16x32_bf16(false, qa0, false, b10, (short)0, s1, false, false);
      s1 = __builtin_amdgcn_wmma_f32_16x16x32_bf16(false, qa1, false, b11, (short)0, s1, false, false);
    }

    // scale + causal mask + online softmax (row stats across the 16 lanes of each half)
#pragma unroll
    for (int r = 0; r < 8; ++r) {
      int qrow = q0 + rhi * 8 + r;
      float a = (kb + rlo      <= qrow) ? s0[r] * scale : -1e30f;
      float b = (kb + 16 + rlo <= qrow) ? s1[r] * scale : -1e30f;
      float mx = fmaxf(a, b);
#pragma unroll
      for (int off = 1; off < 16; off <<= 1) mx = fmaxf(mx, __shfl_xor(mx, off, 32));
      float mn = fmaxf(m[r], mx);
      float resc = __expf(m[r] - mn);
      m[r] = mn;
      float p0 = __expf(a - mn), p1 = __expf(b - mn);
      float rs = p0 + p1;
#pragma unroll
      for (int off = 1; off < 16; off <<= 1) rs += __shfl_xor(rs, off, 32);
      l[r] = l[r] * resc + rs;
#pragma unroll
      for (int c = 0; c < 4; ++c) acc[c][r] *= resc;
      Ps[w][rhi * 8 + r][rlo]      = f2bf(p0);
      Ps[w][rhi * 8 + r][16 + rlo] = f2bf(p1);
    }

    // O += P(16x32) @ V(32x64): four 16-wide hd chunks
    v16bf pf = ld_frag(&Ps[w][rlo][rhi * 16], &Ps[w][rlo][rhi * 16 + 8]);
#pragma unroll
    for (int c = 0; c < 4; ++c) {
      v16bf vf = ld_frag(&Vt[w][c * 16 + rlo][rhi * 16], &Vt[w][c * 16 + rlo][rhi * 16 + 8]);
      acc[c] = __builtin_amdgcn_wmma_f32_16x16x32_bf16(false, pf, false, vf, (short)0, acc[c], false, false);
    }
  }

#pragma unroll
  for (int r = 0; r < 8; ++r) {
    int qrow = q0 + rhi * 8 + r;
    float inv = 1.0f / l[r];
#pragma unroll
    for (int c = 0; c < 4; ++c)
      O[(size_t)qrow * QSTR + h * HDIM + c * 16 + rlo] = f2bf(acc[c][r] * inv);
  }
}

// ---------------- orchestration ----------------
extern "C" void kernel_launch(void* const* d_in, const int* in_sizes, int n_in,
                              void* d_out, int out_size, void* d_ws, size_t ws_size,
                              hipStream_t stream) {
  const float* embed  = (const float*)d_in[0];
  const float* ln1    = (const float*)d_in[1];
  const float* qw     = (const float*)d_in[2];
  const float* qbias  = (const float*)d_in[3];
  const float* kw     = (const float*)d_in[4];
  const float* kbias  = (const float*)d_in[5];
  const float* vw     = (const float*)d_in[6];
  const float* vbias  = (const float*)d_in[7];
  const float* ow     = (const float*)d_in[8];
  const float* ln2    = (const float*)d_in[9];
  const float* gw     = (const float*)d_in[10];
  const float* uw     = (const float*)d_in[11];
  const float* dw     = (const float*)d_in[12];
  const float* norm_w = (const float*)d_in[13];
  const float* lm_w   = (const float*)d_in[14];
  const int*   ids    = (const int*)d_in[15];
  float* out = (float*)d_out;

  // ---- workspace carve-up ----
  char* p = (char*)d_ws;
  auto alloc = [&](size_t bytes) -> void* {
    void* r = (void*)p;
    p += (bytes + 255) & ~(size_t)255;
    return r;
  };
  float*  h    = (float*)alloc((size_t)SEQ * DM * 4);
  __bf16* xb   = (__bf16*)alloc((size_t)SEQ * DM * 2);
  float*  q    = (float*)alloc((size_t)SEQ * QSTR * 4);
  float*  k    = (float*)alloc((size_t)SEQ * KSTR * 4);
  float*  v    = (float*)alloc((size_t)SEQ * KSTR * 4);
  __bf16* qb16 = (__bf16*)alloc((size_t)SEQ * QSTR * 2);
  __bf16* kb16 = (__bf16*)alloc((size_t)SEQ * KSTR * 2);
  __bf16* vb16 = (__bf16*)alloc((size_t)SEQ * KSTR * 2);
  __bf16* aob  = (__bf16*)alloc((size_t)SEQ * QSTR * 2);
  float*  gb   = (float*)alloc((size_t)SEQ * FF * 4);
  float*  ub   = (float*)alloc((size_t)SEQ * FF * 4);
  __bf16* gh   = (__bf16*)alloc((size_t)SEQ * FF * 2);
  __bf16* qwT  = (__bf16*)alloc((size_t)NL * QSTR * DM * 2);
  __bf16* kwT  = (__bf16*)alloc((size_t)NL * KSTR * DM * 2);
  __bf16* vwT  = (__bf16*)alloc((size_t)NL * KSTR * DM * 2);
  __bf16* owT  = (__bf16*)alloc((size_t)NL * DM * QSTR * 2);
  __bf16* gwT  = (__bf16*)alloc((size_t)NL * FF * DM * 2);
  __bf16* uwT  = (__bf16*)alloc((size_t)NL * FF * DM * 2);
  __bf16* dwT  = (__bf16*)alloc((size_t)NL * DM * FF * 2);
  __bf16* lmT  = (__bf16*)alloc((size_t)VOC * DM * 2);

  auto cgrid = [](size_t n) -> unsigned { return (unsigned)((n + 255) / 256); };

  // ---- one-shot weight convert + transpose to bf16 [N][K] ----
  for (int l = 0; l < NL; ++l) {
    convT_kernel<<<cgrid((size_t)DM * QSTR), 256, 0, stream>>>(qw + (size_t)l * DM * QSTR, qwT + (size_t)l * QSTR * DM, DM, QSTR);
    convT_kernel<<<cgrid((size_t)DM * KSTR), 256, 0, stream>>>(kw + (size_t)l * DM * KSTR, kwT + (size_t)l * KSTR * DM, DM, KSTR);
    convT_kernel<<<cgrid((size_t)DM * KSTR), 256, 0, stream>>>(vw + (size_t)l * DM * KSTR, vwT + (size_t)l * KSTR * DM, DM, KSTR);
    convT_kernel<<<cgrid((size_t)QSTR * DM), 256, 0, stream>>>(ow + (size_t)l * QSTR * DM, owT + (size_t)l * DM * QSTR, QSTR, DM);
    convT_kernel<<<cgrid((size_t)DM * FF), 256, 0, stream>>>(gw + (size_t)l * DM * FF, gwT + (size_t)l * FF * DM, DM, FF);
    convT_kernel<<<cgrid((size_t)DM * FF), 256, 0, stream>>>(uw + (size_t)l * DM * FF, uwT + (size_t)l * FF * DM, DM, FF);
    convT_kernel<<<cgrid((size_t)FF * DM), 256, 0, stream>>>(dw + (size_t)l * FF * DM, dwT + (size_t)l * DM * FF, FF, DM);
  }
  convT_kernel<<<cgrid((size_t)DM * VOC), 256, 0, stream>>>(lm_w, lmT, DM, VOC);

  gather_kernel<<<SEQ * DM / 256, 256, 0, stream>>>(embed, ids, h);

  for (int l = 0; l < NL; ++l) {
    rmsnorm_kernel<<<SEQ, 256, 0, stream>>>(h, ln1 + (size_t)l * DM, xb);

    gemm_bf16_wmma<<<dim3(QSTR / 64, SEQ / 128), 256, 0, stream>>>(
        xb, qwT + (size_t)l * QSTR * DM, qbias + (size_t)l * QSTR, nullptr, q, SEQ, QSTR, DM);
    gemm_bf16_wmma<<<dim3(KSTR / 64, SEQ / 128), 256, 0, stream>>>(
        xb, kwT + (size_t)l * KSTR * DM, kbias + (size_t)l * KSTR, nullptr, k, SEQ, KSTR, DM);
    gemm_bf16_wmma<<<dim3(KSTR / 64, SEQ / 128), 256, 0, stream>>>(
        xb, vwT + (size_t)l * KSTR * DM, vbias + (size_t)l * KSTR, nullptr, v, SEQ, KSTR, DM);

    rope_bf16_kernel<<<SEQ * NHQ * 32 / 256, 256, 0, stream>>>(q, qb16, NHQ);
    rope_bf16_kernel<<<SEQ * NKV * 32 / 256, 256, 0, stream>>>(k, kb16, NKV);
    conv_kernel<<<SEQ * KSTR / 256, 256, 0, stream>>>(v, vb16, SEQ * KSTR);

    attn_kernel<<<dim3(SEQ / 64, NHQ), 128, 0, stream>>>(qb16, kb16, vb16, aob);

    // h = h + ao @ ow
    gemm_bf16_wmma<<<dim3(DM / 64, SEQ / 128), 256, 0, stream>>>(
        aob, owT + (size_t)l * DM * QSTR, nullptr, h, h, SEQ, DM, QSTR);

    rmsnorm_kernel<<<SEQ, 256, 0, stream>>>(h, ln2 + (size_t)l * DM, xb);

    gemm_bf16_wmma<<<dim3(FF / 64, SEQ / 128), 256, 0, stream>>>(
        xb, gwT + (size_t)l * FF * DM, nullptr, nullptr, gb, SEQ, FF, DM);
    gemm_bf16_wmma<<<dim3(FF / 64, SEQ / 128), 256, 0, stream>>>(
        xb, uwT + (size_t)l * FF * DM, nullptr, nullptr, ub, SEQ, FF, DM);

    silu_mul_kernel<<<SEQ * FF / 256, 256, 0, stream>>>(gb, ub, gh, SEQ * FF);

    // h = h + (silu(g)*u) @ dw
    gemm_bf16_wmma<<<dim3(DM / 64, SEQ / 128), 256, 0, stream>>>(
        gh, dwT + (size_t)l * DM * FF, nullptr, h, h, SEQ, DM, FF);
  }

  rmsnorm_kernel<<<SEQ, 256, 0, stream>>>(h, norm_w, xb);
  gemm_bf16_wmma<<<dim3(VOC / 64, SEQ / 128), 256, 0, stream>>>(
      xb, lmT, nullptr, nullptr, out, SEQ, VOC, DM);
}
